// SpatialLocalPooling_12936441495847
// MI455X (gfx1250) — matrix operand order repaired and checked by reading, not compiled, not run on hardware
//
#include <hip/hip_runtime.h>

#define NPTS 50000
#define KNN 16
#define CIN 128
#define COUT 128
#define CH 256
#define BN_EPS 1e-5f
#define MTILES (NPTS / 16) // 3125, exact
#define PBATCH 8           // points per attention block (50000/8 = 6250, exact)

typedef __attribute__((ext_vector_type(16))) __bf16 v16bf;
typedef __attribute__((ext_vector_type(8)))  float  v8f;
typedef __attribute__((ext_vector_type(4)))  unsigned int u32x4;

__device__ __forceinline__ v8f wmma_bf16(v16bf a, v16bf b, v8f c) {
    // D = A(16x32 bf16) * B(32x16 bf16) + C(16x16 f32)
    return __builtin_amdgcn_wmma_f32_16x16x32_bf16(
        false, a, false, b, (short)0, c, false, false);
}

// A fragment: 16x32 bf16 tile, row-major source with leading dim `ld` (elems).
// lane<16: M=lane, K in {kk..kk+7, kk+16..kk+23}; lane>=16: M=lane-16, K+8.
__device__ __forceinline__ v16bf frag_a(const __bf16* base, int ld, int kk, int lane) {
    const int m  = lane & 15;
    const int k0 = kk + ((lane & 16) ? 8 : 0);
    const __bf16* p = base + m * ld + k0;
    union { v16bf v; u32x4 q[2]; } u;
    u.q[0] = *(const u32x4*)(p);
    u.q[1] = *(const u32x4*)(p + 16);
    return u.v;
}

// B fragment: 32x16 bf16 tile. Source is "B^T" row-major: row n holds K contiguous.
// lane<16: N=lane, K in {kk..kk+7, kk+16..kk+23}; lane>=16: N=lane-16, K+8.
__device__ __forceinline__ v16bf frag_b(const __bf16* base, int ld, int kk, int lane) {
    const int n  = lane & 15;
    const int k0 = kk + ((lane & 16) ? 8 : 0);
    const __bf16* p = base + n * ld + k0;
    union { v16bf v; u32x4 q[2]; } u;
    u.q[0] = *(const u32x4*)(p);
    u.q[1] = *(const u32x4*)(p + 16);
    return u.v;
}

// ---- 1) convert weights/features to bf16, zero counts ----------------------
__global__ void k_prep(const float* __restrict__ features, const float* __restrict__ W1,
                       const float* __restrict__ Wp, const float* __restrict__ Wd,
                       __bf16* fb, __bf16* W1b, __bf16* Wpb, __bf16* Wdb,
                       int* counts) {
    int i = blockIdx.x * blockDim.x + threadIdx.x;
    if (i < NPTS * CIN) fb[i]  = (__bf16)features[i];
    if (i < CH * CIN)   W1b[i] = (__bf16)W1[i];
    if (i < COUT * CH)  Wpb[i] = (__bf16)Wp[i];
    if (i < COUT * COUT) Wdb[i] = (__bf16)Wd[i];
    if (i < NPTS) counts[i] = 0;
}

// ---- 2) neighbor histogram (integer atomics: deterministic) ----------------
__global__ void k_hist(const int* __restrict__ nbr, int* counts) {
    int i = blockIdx.x * blockDim.x + threadIdx.x;
    if (i < NPTS * KNN) atomicAdd(&counts[nbr[i]], 1);
}

// ---- 3) x = relu(features @ W1^T + b1) -> bf16 (N x 256) -------------------
__global__ void k_gemm1(const __bf16* __restrict__ fb, const __bf16* __restrict__ W1b,
                        const float* __restrict__ b1, __bf16* xb) {
    const int mt   = blockIdx.x;            // 3125 tiles of 16 rows
    const int wave = threadIdx.x >> 5;      // 8 waves
    const int lane = threadIdx.x & 31;
    const __bf16* A = fb + (size_t)mt * 16 * CIN;
    for (int ct = wave * 2; ct < wave * 2 + 2; ++ct) {   // 16 col-tiles, 2/wave
        const int colbase = ct * 16;
        const __bf16* B = W1b + colbase * CIN;
        v8f c = {0.f, 0.f, 0.f, 0.f, 0.f, 0.f, 0.f, 0.f};
#pragma unroll
        for (int kk = 0; kk < CIN; kk += 32)
            c = wmma_bf16(frag_a(A, CIN, kk, lane), frag_b(B, CIN, kk, lane), c);
        const int col  = colbase + (lane & 15);
        const float bv = b1[col];
        const int m0   = (lane & 16) ? 8 : 0;
#pragma unroll
        for (int r = 0; r < 8; ++r) {
            float v = c[r] + bv;
            xb[(size_t)(mt * 16 + m0 + r) * CH + col] = (__bf16)(v > 0.f ? v : 0.f);
        }
    }
}

// ---- 4) count-weighted stats, deterministic partials -----------------------
__global__ void k_stats(const __bf16* __restrict__ xb, const int* __restrict__ counts,
                        float* partial) {
    const int c = threadIdx.x;              // 256 channels
    float s = 0.f, ss = 0.f;
    for (int r = blockIdx.x; r < NPTS; r += gridDim.x) {
        float cnt = (float)counts[r];
        if (cnt != 0.f) {
            float v = (float)xb[(size_t)r * CH + c];
            s  += cnt * v;
            ss += cnt * v * v;
        }
    }
    partial[(size_t)blockIdx.x * 2 * CH + c]      = s;
    partial[(size_t)blockIdx.x * 2 * CH + CH + c] = ss;
}

// ---- 5) reduce partials -> BN affine: a = gamma*rsqrt(var+eps), b = beta-mean*a
__global__ void k_bncoef(const float* __restrict__ partial, const float* __restrict__ gamma,
                         const float* __restrict__ beta, float* ab, int nblk) {
    const int c = threadIdx.x;
    float s = 0.f, ss = 0.f;
    for (int b = 0; b < nblk; ++b) {
        s  += partial[(size_t)b * 2 * CH + c];
        ss += partial[(size_t)b * 2 * CH + CH + c];
    }
    const float inv  = 1.0f / (float)(NPTS * KNN);
    const float mean = s * inv;
    const float var  = ss * inv - mean * mean;
    const float a    = gamma[c] * rsqrtf(var + BN_EPS);
    ab[c]      = a;
    ab[CH + c] = beta[c] - mean * a;
}

// ---- 6) fold BN into Ws: Ws'[o,c] = Ws[o,c]*a_c ; bias_s[o] = sum_c b_c*Ws[o,c]
__global__ void k_fold(const float* __restrict__ Ws, const float* __restrict__ ab,
                       __bf16* Wsb, float* bias_s) {
    __shared__ float red[CH];
    const int o = blockIdx.x, c = threadIdx.x;
    const float w = Ws[(size_t)o * CH + c];
    Wsb[(size_t)o * CH + c] = (__bf16)(w * ab[c]);
    red[c] = w * ab[CH + c];
    __syncthreads();
    for (int s = CH / 2; s > 0; s >>= 1) {
        if (c < s) red[c] += red[c + s];
        __syncthreads();
    }
    if (c == 0) bias_s[o] = red[0];
}

// ---- 7) attention: async-gather PBATCH points, WMMA score GEMM with
//         register-resident B fragments, softmax over K, weighted pool -------
__global__ void __launch_bounds__(256, 1)
k_attn(const __bf16* __restrict__ xb, const int* __restrict__ nbr,
       const __bf16* __restrict__ Wsb, const float* __restrict__ bias_s,
       const float* __restrict__ ab, __bf16* featb) {
    __shared__ __attribute__((aligned(16))) __bf16 g[PBATCH * 16 * CH]; // 64 KB
    const int n0 = blockIdx.x * PBATCH;
    const int t  = threadIdx.x;

    // Async gather: PBATCH*16 rows x 512B = 4096 16B-chunks, 16 per thread.
    // global_load_async_to_lds_b128: per-lane global addr -> per-lane LDS addr,
    // tracked by ASYNCcnt (no VGPR round trip).
#pragma unroll
    for (int i = 0; i < 16; ++i) {
        const int chunk = t + 256 * i;
        const int p  = chunk >> 9;          // 512 chunks per point
        const int k  = (chunk >> 5) & 15;   // 32 chunks per row
        const int e0 = (chunk & 31) * 8;    // bf16 elements
        const int src = nbr[(n0 + p) * KNN + k];
        const __bf16* gp = xb + (size_t)src * CH + e0;
        const unsigned lds = (unsigned)(uintptr_t)(&g[(p * 16 + k) * CH + e0]);
        asm volatile("global_load_async_to_lds_b128 %0, %1, off"
                     :: "v"(lds), "v"(gp) : "memory");
    }
    asm volatile("s_wait_asynccnt 0x0" ::: "memory");
    __syncthreads();

    const int wave = t >> 5, lane = t & 31;
    const int colsel = lane & 15;
    const int m0 = (lane & 16) ? 8 : 0;
    const int col0 = wave * 32 + colsel;        // col-tile 2*wave
    const int col1 = wave * 32 + 16 + colsel;   // col-tile 2*wave+1

    // Pin this wave's B fragments (2 col-tiles x 8 k-steps) in registers:
    // reused across all PBATCH points -> 8x less Ws' traffic from L2.
    v16bf Bf0[8], Bf1[8];
#pragma unroll
    for (int s = 0; s < 8; ++s) {
        Bf0[s] = frag_b(Wsb + (wave * 2) * 16 * CH,     CH, s * 32, lane);
        Bf1[s] = frag_b(Wsb + (wave * 2 + 1) * 16 * CH, CH, s * 32, lane);
    }
    const float bs0 = bias_s[col0], bs1 = bias_s[col1];
    const float a0 = ab[col0], b0 = ab[CH + col0];
    const float a1 = ab[col1], b1v = ab[CH + col1];

    for (int p = 0; p < PBATCH; ++p) {
        const __bf16* A = &g[p * 16 * CH];
        v8f c0, c1;
#pragma unroll
        for (int r = 0; r < 8; ++r) { c0[r] = bs0; c1[r] = bs1; }
#pragma unroll
        for (int s = 0; s < 8; ++s) {
            v16bf a = frag_a(A, CH, s * 32, lane);  // shared A frag for both tiles
            c0 = wmma_bf16(a, Bf0[s], c0);
            c1 = wmma_bf16(a, Bf1[s], c1);
        }
        // softmax over the 16 K-rows of each column (lane pair L / L+16),
        // then feat_c = a_c * sum_k p_k * x_k  + b_c   (softmax sums to 1)
#pragma unroll
        for (int tile = 0; tile < 2; ++tile) {
            v8f c = tile ? c1 : c0;
            const int col = tile ? col1 : col0;
            float m = c[0];
#pragma unroll
            for (int r = 1; r < 8; ++r) m = fmaxf(m, c[r]);
            m = fmaxf(m, __shfl_xor(m, 16, 32));
            float pr[8], sum = 0.f;
#pragma unroll
            for (int r = 0; r < 8; ++r) { pr[r] = __expf(c[r] - m); sum += pr[r]; }
            sum += __shfl_xor(sum, 16, 32);
            const float invs = 1.0f / sum;
            float acc = 0.f;
#pragma unroll
            for (int r = 0; r < 8; ++r)
                acc += pr[r] * (float)A[(m0 + r) * CH + col];
            acc *= invs;
            acc += __shfl_xor(acc, 16, 32);
            if (lane < 16) {
                const float av = tile ? a1 : a0;
                const float bv = tile ? b1v : b0;
                featb[(size_t)(n0 + p) * CH + col] = (__bf16)(av * acc + bv);
            }
        }
    }
}

// ---- 8) embedding = feat @ Wp^T + bp  (f32 out + bf16 copy) ----------------
__global__ void k_gemm_emb(const __bf16* __restrict__ featb, const __bf16* __restrict__ Wpb,
                           const float* __restrict__ bp, float* out_emb, __bf16* embb) {
    const int mt = blockIdx.x;
    const int wave = threadIdx.x >> 5, lane = threadIdx.x & 31;
    const int colbase = wave * 16;                  // 8 col-tiles, 1/wave
    const __bf16* A = featb + (size_t)mt * 16 * CH;
    const __bf16* B = Wpb + colbase * CH;
    v8f c = {0.f, 0.f, 0.f, 0.f, 0.f, 0.f, 0.f, 0.f};
#pragma unroll
    for (int kk = 0; kk < CH; kk += 32)
        c = wmma_bf16(frag_a(A, CH, kk, lane), frag_b(B, CH, kk, lane), c);
    const int col = colbase + (lane & 15);
    const float bv = bp[col];
    const int m0 = (lane & 16) ? 8 : 0;
#pragma unroll
    for (int r = 0; r < 8; ++r) {
        float v = c[r] + bv;
        size_t o = (size_t)(mt * 16 + m0 + r) * COUT + col;
        out_emb[o] = v;
        embb[o] = (__bf16)v;
    }
}

// ---- 9) reconstructed = relu(emb @ Wd^T + bd) ------------------------------
__global__ void k_gemm_rec(const __bf16* __restrict__ embb, const __bf16* __restrict__ Wdb,
                           const float* __restrict__ bd, float* out_rec) {
    const int mt = blockIdx.x;
    const int wave = threadIdx.x >> 5, lane = threadIdx.x & 31;
    const int colbase = wave * 16;                  // 8 col-tiles of CIN=128
    const __bf16* A = embb + (size_t)mt * 16 * COUT;
    const __bf16* B = Wdb + colbase * COUT;
    v8f c = {0.f, 0.f, 0.f, 0.f, 0.f, 0.f, 0.f, 0.f};
#pragma unroll
    for (int kk = 0; kk < COUT; kk += 32)
        c = wmma_bf16(frag_a(A, COUT, kk, lane), frag_b(B, COUT, kk, lane), c);
    const int col = colbase + (lane & 15);
    const float bv = bd[col];
    const int m0 = (lane & 16) ? 8 : 0;
#pragma unroll
    for (int r = 0; r < 8; ++r) {
        float v = c[r] + bv;
        out_rec[(size_t)(mt * 16 + m0 + r) * CIN + col] = v > 0.f ? v : 0.f;
    }
}

extern "C" void kernel_launch(void* const* d_in, const int* in_sizes, int n_in,
                              void* d_out, int out_size, void* d_ws, size_t ws_size,
                              hipStream_t stream) {
    (void)in_sizes; (void)n_in; (void)out_size; (void)ws_size;
    const float* features = (const float*)d_in[0];
    const int*   nbr      = (const int*)d_in[1];
    const float* W1       = (const float*)d_in[2];
    const float* b1       = (const float*)d_in[3];
    const float* gamma    = (const float*)d_in[4];
    const float* beta     = (const float*)d_in[5];
    const float* Ws       = (const float*)d_in[6];
    const float* Wp       = (const float*)d_in[7];
    const float* bp       = (const float*)d_in[8];
    const float* Wd       = (const float*)d_in[9];
    const float* bd       = (const float*)d_in[10];

    float* out_rec = (float*)d_out;                      // (N, C_IN)
    float* out_emb = out_rec + (size_t)NPTS * CIN;       // (N, C_OUT)

    char* w = (char*)d_ws;
    auto take = [&](size_t bytes) -> char* {
        char* p = w; w += (bytes + 255) & ~(size_t)255; return p;
    };
    __bf16* fb     = (__bf16*)take((size_t)NPTS * CIN * 2);   // features bf16
    __bf16* xb     = (__bf16*)take((size_t)NPTS * CH * 2);    // relu(mlp1) bf16
    __bf16* featb  = (__bf16*)take((size_t)NPTS * CH * 2);    // pooled feat bf16
    __bf16* embb   = (__bf16*)take((size_t)NPTS * COUT * 2);  // embedding bf16
    int*    counts = (int*)take((size_t)NPTS * 4);
    float*  partial= (float*)take((size_t)256 * 2 * CH * 4);  // stats partials
    float*  ab     = (float*)take((size_t)2 * CH * 4);        // BN affine a,b
    __bf16* W1b    = (__bf16*)take((size_t)CH * CIN * 2);
    __bf16* Wsb    = (__bf16*)take((size_t)CH * CH * 2);      // BN-folded Ws
    float*  bias_s = (float*)take((size_t)CH * 4);
    __bf16* Wpb    = (__bf16*)take((size_t)COUT * CH * 2);
    __bf16* Wdb    = (__bf16*)take((size_t)COUT * COUT * 2);

    const int STATS_BLKS = 256;
    k_prep<<<(NPTS * CIN + 255) / 256, 256, 0, stream>>>(features, W1, Wp, Wd,
                                                         fb, W1b, Wpb, Wdb, counts);
    k_hist<<<(NPTS * KNN + 255) / 256, 256, 0, stream>>>(nbr, counts);
    k_gemm1<<<MTILES, 256, 0, stream>>>(fb, W1b, b1, xb);
    k_stats<<<STATS_BLKS, CH, 0, stream>>>(xb, counts, partial);
    k_bncoef<<<1, CH, 0, stream>>>(partial, gamma, beta, ab, STATS_BLKS);
    k_fold<<<CH, CH, 0, stream>>>(Ws, ab, Wsb, bias_s);
    k_attn<<<NPTS / PBATCH, 256, 0, stream>>>(xb, nbr, Wsb, bias_s, ab, featb);
    k_gemm_emb<<<MTILES, 256, 0, stream>>>(featb, Wpb, bp, out_emb, embb);
    k_gemm_rec<<<MTILES, 256, 0, stream>>>(embb, Wdb, bd, out_rec);
}
